// RBFKernel_65481071395625
// MI455X (gfx1250) — compile-verified, matrix operand order checked by reading, not run on hardware
//
#include <hip/hip_runtime.h>
#include <stdint.h>

typedef __attribute__((ext_vector_type(16))) __bf16    v16bf;
typedef __attribute__((ext_vector_type(8)))  float     v8f;
typedef __attribute__((ext_vector_type(4)))  uint32_t  u32x4;
typedef int v4i __attribute__((vector_size(16)));   // matches builtin's expected pointee

#define NROWS 8192
#define DDIM  256
#define GAMMA 1.0f
#define BK    32
#define ROWSTR 40                 // padded LDS row stride (bf16): 64B data + 16B pad
#define TILE  (128 * ROWSTR)      // one 128x32 bf16 tile (padded) in LDS
#define AS1 __attribute__((address_space(1)))
#define AS3 __attribute__((address_space(3)))

#if __has_builtin(__builtin_amdgcn_global_load_async_to_lds_b128)
#define USE_ASYNC 1
#else
#define USE_ASYNC 0
#endif

// ---------------- prep: norms + bf16 hi/lo split ----------------
__global__ void __launch_bounds__(256)
rbf_prep(const float* __restrict__ x, const float* __restrict__ y,
         __bf16* __restrict__ xh, __bf16* __restrict__ xl,
         __bf16* __restrict__ yh, __bf16* __restrict__ yl,
         float* __restrict__ xs, float* __restrict__ ys)
{
    const int row = blockIdx.x;
    const int t   = threadIdx.x;
    const float* src = blockIdx.y ? y  : x;
    __bf16* hi       = blockIdx.y ? yh : xh;
    __bf16* lo       = blockIdx.y ? yl : xl;
    float*  sq       = blockIdx.y ? ys : xs;

    float f = src[(size_t)row * DDIM + t];
    __bf16 h = (__bf16)f;
    __bf16 l = (__bf16)(f - (float)h);
    hi[(size_t)row * DDIM + t] = h;
    lo[(size_t)row * DDIM + t] = l;

    __shared__ float red[256];
    red[t] = f * f;
    __syncthreads();
    #pragma unroll
    for (int s = 128; s > 0; s >>= 1) {
        if (t < s) red[t] += red[t + s];
        __syncthreads();
    }
    if (t == 0) sq[row] = red[0];
}

// ---------------- main: WMMA GEMM + fused RBF epilogue ----------------
union Frag { v16bf bf; u32x4 q[2]; };

__device__ inline v16bf ld_frag(const __bf16* rowbase, int half) {
    // 16x32 bf16 A/B fragment: lanes 0-15 hold K={0..7,16..23}, lanes 16-31 K={8..15,24..31}
    const u32x4* p = (const u32x4*)rowbase;
    Frag f;
    f.q[0] = p[half];       // K = half*8 .. half*8+7
    f.q[1] = p[2 + half];   // K = 16 + half*8 ..
    return f.bf;
}

__device__ inline void copy16(const __bf16* gsrc, __bf16* ldst) {
#if USE_ASYNC
    // gfx1250 async DMA: global -> LDS, tracked by ASYNCcnt, no VGPR staging
    __builtin_amdgcn_global_load_async_to_lds_b128(
        (AS1 v4i*)gsrc, (AS3 v4i*)ldst, 0, 0);
#else
    *(u32x4*)ldst = *(const u32x4*)gsrc;
#endif
}

__device__ inline void wait_async_lds() {
#if USE_ASYNC
#if __has_builtin(__builtin_amdgcn_s_wait_asynccnt)
    __builtin_amdgcn_s_wait_asynccnt(0);
#else
    asm volatile("s_wait_asynccnt 0x0" ::: "memory");
#endif
#endif
}

__device__ inline void stage_tiles(const __bf16* __restrict__ xh,
                                   const __bf16* __restrict__ xl,
                                   const __bf16* __restrict__ yh,
                                   const __bf16* __restrict__ yl,
                                   __bf16* lbuf, int m0, int n0, int k0, int tid)
{
    #pragma unroll
    for (int c = 0; c < 2; ++c) {
        const int chunk = tid + c * 256;     // 0..511
        const int row   = chunk >> 2;        // 0..127
        const int off   = (chunk & 3) * 8;   // bf16 elems within the 32-wide chunk
        const size_t gx = (size_t)(m0 + row) * DDIM + k0 + off;
        const size_t gy = (size_t)(n0 + row) * DDIM + k0 + off;
        const int    lo = row * ROWSTR + off;
        copy16(xh + gx, lbuf + 0 * TILE + lo);
        copy16(xl + gx, lbuf + 1 * TILE + lo);
        copy16(yh + gy, lbuf + 2 * TILE + lo);
        copy16(yl + gy, lbuf + 3 * TILE + lo);
    }
}

__global__ void __launch_bounds__(256)
rbf_main(const __bf16* __restrict__ xh, const __bf16* __restrict__ xl,
         const __bf16* __restrict__ yh, const __bf16* __restrict__ yl,
         const float* __restrict__ xs, const float* __restrict__ ys,
         float* __restrict__ out)
{
    __shared__ __bf16 lds[2 * 4 * TILE];   // double-buffered: 2 x (xh,xl,yh,yl) tiles

    const int tid    = threadIdx.x;
    const int lane   = tid & 31;
    const int wid    = tid >> 5;
    const int wave_m = wid & 3;   // 4 waves over M (32 rows each)
    const int wave_n = wid >> 2;  // 2 waves over N (64 cols each)
    const int half   = (lane >> 4) & 1;
    const int l16    = lane & 15;

    const int m0 = blockIdx.y * 128;
    const int n0 = blockIdx.x * 128;

    v8f acc[2][4] = {};

    // prologue: fill buffer 0
    stage_tiles(xh, xl, yh, yl, lds, m0, n0, 0, tid);
    wait_async_lds();
    __syncthreads();

    for (int kc = 0; kc < DDIM / BK; ++kc) {
        __bf16* cur = lds + ((kc & 1) ? 4 * TILE : 0);
        __bf16* nxt = lds + ((kc & 1) ? 0 : 4 * TILE);

        // kick off async copy of the next K-chunk while we compute this one
        if (kc + 1 < DDIM / BK)
            stage_tiles(xh, xl, yh, yl, nxt, m0, n0, (kc + 1) * BK, tid);

        v16bf ah[2], al[2], bh[4], bl[4];
        #pragma unroll
        for (int i = 0; i < 2; ++i) {
            const int r = wave_m * 32 + i * 16 + l16;
            ah[i] = ld_frag(cur + 0 * TILE + r * ROWSTR, half);
            al[i] = ld_frag(cur + 1 * TILE + r * ROWSTR, half);
        }
        #pragma unroll
        for (int j = 0; j < 4; ++j) {
            const int r = wave_n * 64 + j * 16 + l16;
            bh[j] = ld_frag(cur + 2 * TILE + r * ROWSTR, half);
            bl[j] = ld_frag(cur + 3 * TILE + r * ROWSTR, half);
        }

        // 3-term bf16 split: hi*hi + hi*lo + lo*hi  (~fp32 accuracy)
        #pragma unroll
        for (int i = 0; i < 2; ++i) {
            #pragma unroll
            for (int j = 0; j < 4; ++j) {
                acc[i][j] = __builtin_amdgcn_wmma_f32_16x16x32_bf16(
                    false, ah[i], false, bh[j], (short)0, acc[i][j], false, false);
                acc[i][j] = __builtin_amdgcn_wmma_f32_16x16x32_bf16(
                    false, ah[i], false, bl[j], (short)0, acc[i][j], false, false);
                acc[i][j] = __builtin_amdgcn_wmma_f32_16x16x32_bf16(
                    false, al[i], false, bh[j], (short)0, acc[i][j], false, false);
            }
        }

        // make the freshly staged buffer visible to all waves before next iter
        wait_async_lds();
        __syncthreads();
    }

    // Fused epilogue: out = exp(-gamma * (x2 - 2*cross + y2)); NT stores (write-once
    // 256MB stream; keep the 192MB L2 for the GEMM operands)
    #pragma unroll
    for (int i = 0; i < 2; ++i) {
        #pragma unroll
        for (int j = 0; j < 4; ++j) {
            const int mb  = m0 + wave_m * 32 + i * 16 + half * 8;
            const int nb  = n0 + wave_n * 64 + j * 16 + l16;
            const float y2 = ys[nb];
            #pragma unroll
            for (int v = 0; v < 8; ++v) {
                const int m = mb + v;
                const float d2 = xs[m] - 2.0f * acc[i][j][v] + y2;
                __builtin_nontemporal_store(__expf(-GAMMA * d2),
                                            out + (size_t)m * NROWS + nb);
            }
        }
    }
}

extern "C" void kernel_launch(void* const* d_in, const int* in_sizes, int n_in,
                              void* d_out, int out_size, void* d_ws, size_t ws_size,
                              hipStream_t stream) {
    const float* x = (const float*)d_in[0];
    const float* y = (const float*)d_in[1];
    float* out = (float*)d_out;

    // workspace layout: xh, xl, yh, yl (bf16), then xs, ys (f32) -> ~16.06 MiB
    char* ws = (char*)d_ws;
    __bf16* xh = (__bf16*)ws;
    __bf16* xl = xh + (size_t)NROWS * DDIM;
    __bf16* yh = xl + (size_t)NROWS * DDIM;
    __bf16* yl = yh + (size_t)NROWS * DDIM;
    float*  xs = (float*)(yl + (size_t)NROWS * DDIM);
    float*  ys = xs + NROWS;

    rbf_prep<<<dim3(NROWS, 2), 256, 0, stream>>>(x, y, xh, xl, yh, yl, xs, ys);
    rbf_main<<<dim3(NROWS / 128, NROWS / 128), 256, 0, stream>>>(xh, xl, yh, yl, xs, ys, out);
}
